// CSSMB_25683904430104
// MI455X (gfx1250) — matrix operand order repaired and checked by reading, not compiled, not run on hardware
//
#include <hip/hip_runtime.h>
#include <math.h>

#define Bsz 8
#define Lseq 4096
#define DIm 6
#define DSm 16
#define DCk 4
#define KTOT (Lseq * 3)   // 12288
#define NPAD 32

typedef __bf16 bf16_t;
typedef __attribute__((ext_vector_type(16))) __bf16 v16bf;
typedef __attribute__((ext_vector_type(8)))  float  v8f;

static inline int cdiv(int a, int b) { return (a + b - 1) / b; }

// ---------------------------------------------------------------------------
// 64-point DFT along rows (axis=0: base=line*64, stride=1) or cols (axis=1).
// 24 images of 64x64, contiguous. sign=-1 fwd, +1 inv. scale applied per pass.
// ---------------------------------------------------------------------------
__global__ void k_dft64(const float* __restrict__ in_re, const float* __restrict__ in_im,
                        float* __restrict__ out_re, float* __restrict__ out_im,
                        int axis, float sign, float scale, int has_im, int has_out_im) {
    __shared__ float sre[64];
    __shared__ float sim[64];
    int li = blockIdx.x;
    int t = threadIdx.x;
    size_t base; int stride;
    if (axis == 0) { base = (size_t)li * 64; stride = 1; }
    else { int img = li >> 6, xc = li & 63; base = (size_t)img * 4096 + xc; stride = 64; }
    sre[t] = in_re[base + (size_t)t * stride];
    sim[t] = has_im ? in_im[base + (size_t)t * stride] : 0.f;
    __syncthreads();
    float ar = 0.f, ai = 0.f;
    const float w0 = sign * 6.283185307179586f / 64.f;
    for (int n = 0; n < 64; ++n) {
        float sn, cn;
        __sincosf(w0 * (float)(t * n), &sn, &cn);
        ar += sre[n] * cn - sim[n] * sn;
        ai += sre[n] * sn + sim[n] * cn;
    }
    out_re[base + (size_t)t * stride] = ar * scale;
    if (has_out_im) out_im[base + (size_t)t * stride] = ai * scale;
}

// ---------------------------------------------------------------------------
// convb: 3x3 conv 3->8 + relu, then 3x3 conv 8->3 + residual
// ---------------------------------------------------------------------------
__global__ void k_convb1(const float* __restrict__ in, const float* __restrict__ w,
                         const float* __restrict__ bias, float* __restrict__ out, int total) {
    int i = blockIdx.x * blockDim.x + threadIdx.x;
    if (i >= total) return;                       // B*8*64*64
    int x = i & 63, y = (i >> 6) & 63, oc = (i >> 12) & 7, b = i >> 15;
    float acc = bias[oc];
    for (int ic = 0; ic < 3; ++ic)
        for (int ky = 0; ky < 3; ++ky) {
            int iy = y + ky - 1; if ((unsigned)iy >= 64u) continue;
            for (int kx = 0; kx < 3; ++kx) {
                int ix = x + kx - 1; if ((unsigned)ix >= 64u) continue;
                acc += in[(((size_t)b * 3 + ic) * 64 + iy) * 64 + ix] *
                       w[((oc * 3 + ic) * 3 + ky) * 3 + kx];
            }
        }
    out[i] = fmaxf(acc, 0.f);
}

__global__ void k_convb2(const float* __restrict__ mid, const float* __restrict__ w,
                         const float* __restrict__ bias, const float* __restrict__ res,
                         float* __restrict__ out, int total) {
    int i = blockIdx.x * blockDim.x + threadIdx.x;
    if (i >= total) return;                       // B*3*64*64
    int x = i & 63, y = (i >> 6) & 63, t2 = i >> 12;
    int oc = t2 % 3, b = t2 / 3;
    float acc = bias[oc];
    for (int ic = 0; ic < 8; ++ic)
        for (int ky = 0; ky < 3; ++ky) {
            int iy = y + ky - 1; if ((unsigned)iy >= 64u) continue;
            for (int kx = 0; kx < 3; ++kx) {
                int ix = x + kx - 1; if ((unsigned)ix >= 64u) continue;
                acc += mid[(((size_t)b * 8 + ic) * 64 + iy) * 64 + ix] *
                       w[((oc * 8 + ic) * 3 + ky) * 3 + kx];
            }
        }
    out[i] = res[i] + acc;
}

// LayerNorm over groups of 3 (in place)
__global__ void k_ln(float* __restrict__ t, const float* __restrict__ g,
                     const float* __restrict__ bb, int total) {
    int i = blockIdx.x * blockDim.x + threadIdx.x;
    if (i >= total) return;                       // B*L
    float a0 = t[(size_t)i * 3 + 0], a1 = t[(size_t)i * 3 + 1], a2 = t[(size_t)i * 3 + 2];
    float mu = (a0 + a1 + a2) * (1.f / 3.f);
    float d0 = a0 - mu, d1 = a1 - mu, d2 = a2 - mu;
    float inv = rsqrtf((d0 * d0 + d1 * d1 + d2 * d2) * (1.f / 3.f) + 1e-5f);
    t[(size_t)i * 3 + 0] = d0 * inv * g[0] + bb[0];
    t[(size_t)i * 3 + 1] = d1 * inv * g[1] + bb[1];
    t[(size_t)i * 3 + 2] = d2 * inv * g[2] + bb[2];
}

// ---------------------------------------------------------------------------
// Mamba
// ---------------------------------------------------------------------------
__global__ void k_xz(const float* __restrict__ u, const float* __restrict__ in_w,
                     float* __restrict__ xz, int total) {
    int i = blockIdx.x * blockDim.x + threadIdx.x;
    if (i >= total) return;                       // B*L
    float u0 = u[(size_t)i * 3], u1 = u[(size_t)i * 3 + 1], u2 = u[(size_t)i * 3 + 2];
#pragma unroll
    for (int r = 0; r < 12; ++r)
        xz[(size_t)i * 12 + r] = in_w[r * 3] * u0 + in_w[r * 3 + 1] * u1 + in_w[r * 3 + 2] * u2;
}

__global__ void k_conv_silu(const float* __restrict__ xz, const float* __restrict__ cw,
                            const float* __restrict__ cb, float* __restrict__ xc, int total) {
    int i = blockIdx.x * blockDim.x + threadIdx.x;
    if (i >= total) return;                       // B*L*DI
    int d = i % DIm; int bl = i / DIm; int l = bl % Lseq; int b = bl / Lseq;
    float acc = cb[d];
#pragma unroll
    for (int k = 0; k < DCk; ++k) {
        int ll = l - (DCk - 1) + k;
        if (ll >= 0) acc += cw[d * DCk + k] * xz[((size_t)b * Lseq + ll) * 12 + d];
    }
    xc[(size_t)bl * DIm + d] = acc / (1.f + __expf(-acc));
}

__global__ void k_xdbl_dt(const float* __restrict__ xc, const float* __restrict__ xp_w,
                          const float* __restrict__ dt_w, const float* __restrict__ dt_b,
                          float* __restrict__ xdbl, float* __restrict__ dt, int total) {
    int i = blockIdx.x * blockDim.x + threadIdx.x;
    if (i >= total) return;                       // B*L
    float v[DIm];
#pragma unroll
    for (int d = 0; d < DIm; ++d) v[d] = xc[(size_t)i * DIm + d];
    float x0 = 0.f;
    for (int r = 0; r < 33; ++r) {
        float a = 0.f;
#pragma unroll
        for (int d = 0; d < DIm; ++d) a += xp_w[r * DIm + d] * v[d];
        xdbl[(size_t)i * 33 + r] = a;
        if (r == 0) x0 = a;
    }
#pragma unroll
    for (int d = 0; d < DIm; ++d) {
        float p = x0 * dt_w[d] + dt_b[d];
        dt[(size_t)i * DIm + d] = (p > 20.f) ? p : log1pf(__expf(p));
    }
}

// selective scan: one block per batch, 96 threads = (d,s), shuffle-reduce over 16 lanes
__global__ void k_scan(const float* __restrict__ dt, const float* __restrict__ xdbl,
                       const float* __restrict__ xc, const float* __restrict__ A_log,
                       float* __restrict__ ys) {
    int b = blockIdx.x;
    int t = threadIdx.x;                           // 0..95
    int d = t >> 4, s = t & 15;
    float A = -__expf(A_log[d * DSm + s]);
    float h = 0.f;
    for (int l = 0; l < Lseq; ++l) {
        size_t base = (size_t)b * Lseq + l;
        float dtv = dt[base * DIm + d];
        float Bv  = xdbl[base * 33 + 1 + s];
        float Cv  = xdbl[base * 33 + 17 + s];
        float xv  = xc[base * DIm + d];
        h = h * __expf(dtv * A) + dtv * Bv * xv;
        float p = h * Cv;
        p += __shfl_xor(p, 1, 32);
        p += __shfl_xor(p, 2, 32);
        p += __shfl_xor(p, 4, 32);
        p += __shfl_xor(p, 8, 32);
        if (s == 0) ys[base * DIm + d] = p;
    }
}

__global__ void k_post(const float* __restrict__ ys, const float* __restrict__ xc,
                       const float* __restrict__ xz, const float* __restrict__ Dp,
                       const float* __restrict__ out_w, float* __restrict__ m_out, int total) {
    int i = blockIdx.x * blockDim.x + threadIdx.x;
    if (i >= total) return;                       // B*L
    float y[DIm];
#pragma unroll
    for (int d = 0; d < DIm; ++d) {
        float z = xz[(size_t)i * 12 + 6 + d];
        float sz = z / (1.f + __expf(-z));
        y[d] = (ys[(size_t)i * DIm + d] + Dp[d] * xc[(size_t)i * DIm + d]) * sz;
    }
#pragma unroll
    for (int j = 0; j < 3; ++j) {
        float a = 0.f;
#pragma unroll
        for (int d = 0; d < DIm; ++d) a += y[d] * out_w[j * DIm + d];
        m_out[(size_t)i * 3 + j] = a;
    }
}

// softmax over batch axis: in/out (B, L*3)
__global__ void k_softmax_b(const float* __restrict__ in, float* __restrict__ out, int total) {
    int i = blockIdx.x * blockDim.x + threadIdx.x;
    if (i >= total) return;                       // L*3
    float v[Bsz]; float mx = -1e30f;
#pragma unroll
    for (int b = 0; b < Bsz; ++b) { v[b] = in[(size_t)b * 12288 + i]; mx = fmaxf(mx, v[b]); }
    float s = 0.f;
#pragma unroll
    for (int b = 0; b < Bsz; ++b) { v[b] = __expf(v[b] - mx); s += v[b]; }
    float inv = 1.f / s;
#pragma unroll
    for (int b = 0; b < Bsz; ++b) out[(size_t)b * 12288 + i] = v[b] * inv;
}

// pack activations (B,L,3) into transposed bf16 B-matrix XpT[32][KTOT]
__global__ void k_pack(const float* __restrict__ t, bf16_t* __restrict__ Xp, int total) {
    int i = blockIdx.x * blockDim.x + threadIdx.x;
    if (i >= total) return;                       // 32*KTOT
    int K = i % KTOT; int n = i / KTOT;
    float val = 0.f;
    if (n < 24) {
        int b = n / 3, h = n % 3;
        int l_in = K / 3, kk = K % 3;
        int hh = h + kk - 1;
        if (hh >= 0 && hh < 3) val = t[((size_t)b * Lseq + l_in) * 3 + hh];
    }
    Xp[(size_t)n * KTOT + K] = (bf16_t)val;
}

// ---------------------------------------------------------------------------
// The big conv1 GEMM: Y[4096x32] = W[4096x12288](fp32) x XpT^T(bf16)
// bf16 WMMA, A converted fp32->bf16 in-register; weight read exactly once.
// ---------------------------------------------------------------------------
__global__ __launch_bounds__(128) void k_gemm_wmma(const float* __restrict__ Aw,
                                                   const bf16_t* __restrict__ BT,
                                                   float* __restrict__ Y) {
    const int lane = threadIdx.x & 31;
    const int wave = threadIdx.x >> 5;
    const int m0 = blockIdx.x * 64 + wave * 16;
    const int mrow = lane & 15;
    const int kb = (lane >> 4) << 4;            // A-fragment: lanes>=16 hold K=16..31
    const int kofsB = (lane >> 4) * 16;         // B-fragment: lanes>=16 hold K=16..31
    const float* arow = Aw + (size_t)(m0 + mrow) * KTOT;
    const bf16_t* b0row = BT + (size_t)(lane & 15) * KTOT;
    const bf16_t* b1row = BT + (size_t)((lane & 15) + 16) * KTOT;
    v8f acc0 = {}; v8f acc1 = {};
    for (int k0 = 0; k0 < KTOT; k0 += 32) {
        if (k0 + 512 < KTOT) __builtin_prefetch(arow + k0 + 512, 0, 1);
        const float4 a0 = *(const float4*)(arow + k0 + kb);
        const float4 a1 = *(const float4*)(arow + k0 + kb + 4);
        const float4 a2 = *(const float4*)(arow + k0 + kb + 8);
        const float4 a3 = *(const float4*)(arow + k0 + kb + 12);
        v16bf a;
        a[0]  = (bf16_t)a0.x; a[1]  = (bf16_t)a0.y; a[2]  = (bf16_t)a0.z; a[3]  = (bf16_t)a0.w;
        a[4]  = (bf16_t)a1.x; a[5]  = (bf16_t)a1.y; a[6]  = (bf16_t)a1.z; a[7]  = (bf16_t)a1.w;
        a[8]  = (bf16_t)a2.x; a[9]  = (bf16_t)a2.y; a[10] = (bf16_t)a2.z; a[11] = (bf16_t)a2.w;
        a[12] = (bf16_t)a3.x; a[13] = (bf16_t)a3.y; a[14] = (bf16_t)a3.z; a[15] = (bf16_t)a3.w;
        v16bf b0 = *(const v16bf*)(b0row + k0 + kofsB);
        v16bf b1 = *(const v16bf*)(b1row + k0 + kofsB);
        acc0 = __builtin_amdgcn_wmma_f32_16x16x32_bf16(false, a, false, b0, (short)0, acc0, false, false);
        acc1 = __builtin_amdgcn_wmma_f32_16x16x32_bf16(false, a, false, b1, (short)0, acc1, false, false);
    }
    const int nb = lane & 15;
    const int mo = (lane >> 4) * 8;             // D layout: lanes>=16 hold M=8..15
#pragma unroll
    for (int r = 0; r < 8; ++r) {
        int m = m0 + mo + r;
        Y[(size_t)m * NPAD + nb] = acc0[r];
        Y[(size_t)m * NPAD + 16 + nb] = acc1[r];
    }
}

// amp3/amp4 -> cross = amp3 * softmax_b(amp3)   (note amp3 == pha3 in the reference)
__global__ void k_cross(const float* __restrict__ Y, const float* __restrict__ bias,
                        float* __restrict__ cross, int total) {
    int i = blockIdx.x * blockDim.x + threadIdx.x;
    if (i >= total) return;                       // L*3
    int l = i / 3, h = i % 3;
    float v[Bsz]; float mx = -1e30f;
#pragma unroll
    for (int b = 0; b < Bsz; ++b) { v[b] = Y[(size_t)l * NPAD + b * 3 + h] + bias[l]; mx = fmaxf(mx, v[b]); }
    float s = 0.f; float e[Bsz];
#pragma unroll
    for (int b = 0; b < Bsz; ++b) { e[b] = __expf(v[b] - mx); s += e[b]; }
    float inv = 1.f / s;
#pragma unroll
    for (int b = 0; b < Bsz; ++b) cross[(size_t)b * 12288 + i] = v[b] * e[b] * inv;
}

__global__ void k_compose(const float* __restrict__ Ya, const float* __restrict__ ba,
                          const float* __restrict__ sma, const float* __restrict__ Yp,
                          const float* __restrict__ bp, const float* __restrict__ smp,
                          const float* __restrict__ cross, const float* __restrict__ araw,
                          const float* __restrict__ praw, float* __restrict__ aout,
                          float* __restrict__ pout, int total) {
    int i = blockIdx.x * blockDim.x + threadIdx.x;
    if (i >= total) return;                       // B*12288
    int b = i / 12288; int p = i % 12288; int l = p / 3; int h = p % 3;
    int n = b * 3 + h;
    float a1 = Ya[(size_t)l * NPAD + n] + ba[l];
    float p1 = Yp[(size_t)l * NPAD + n] + bp[l];
    float cr = cross[i];
    aout[i] = a1 * sma[(size_t)b * 12288 + p] + cr + araw[i];
    pout[i] = p1 * smp[(size_t)b * 12288 + p] + cr + praw[i];
}

// ---------------------------------------------------------------------------
static void run_mamba(const float* u, const float* in_w, const float* conv_w,
                      const float* conv_b, const float* xp_w, const float* dt_w,
                      const float* dt_b, const float* A_log, const float* Dp,
                      const float* out_w, float* xz, float* xc, float* xdbl,
                      float* dt, float* ys, float* m_out, hipStream_t stream) {
    const int BL = Bsz * Lseq;
    k_xz<<<cdiv(BL, 256), 256, 0, stream>>>(u, in_w, xz, BL);
    k_conv_silu<<<cdiv(BL * DIm, 256), 256, 0, stream>>>(xz, conv_w, conv_b, xc, BL * DIm);
    k_xdbl_dt<<<cdiv(BL, 256), 256, 0, stream>>>(xc, xp_w, dt_w, dt_b, xdbl, dt, BL);
    k_scan<<<Bsz, 96, 0, stream>>>(dt, xdbl, xc, A_log, ys);
    k_post<<<cdiv(BL, 256), 256, 0, stream>>>(ys, xc, xz, Dp, out_w, m_out, BL);
}

extern "C" void kernel_launch(void* const* d_in, const int* in_sizes, int n_in,
                              void* d_out, int out_size, void* d_ws, size_t ws_size,
                              hipStream_t stream) {
    const float* x      = (const float*)d_in[0];
    const float* cb1_w  = (const float*)d_in[1];
    const float* cb1_b  = (const float*)d_in[2];
    const float* cb2_w  = (const float*)d_in[3];
    const float* cb2_b  = (const float*)d_in[4];
    const float* ln_g   = (const float*)d_in[5];
    const float* ln_b   = (const float*)d_in[6];
    const float* m1p[9]; for (int i = 0; i < 9; ++i) m1p[i] = (const float*)d_in[7 + i];
    const float* m2p[9]; for (int i = 0; i < 9; ++i) m2p[i] = (const float*)d_in[16 + i];
    const float* c11_w  = (const float*)d_in[25];
    const float* c11_b  = (const float*)d_in[26];
    const float* c12_w  = (const float*)d_in[27];
    const float* c12_b  = (const float*)d_in[28];
    const float* cr1_w  = (const float*)d_in[29];
    const float* cr1_b  = (const float*)d_in[30];

    float* ws = (float*)d_ws;
    size_t off = 0;
    auto alloc = [&](size_t n) { float* p = ws + off; off += n; return p; };
    const size_t IMG = (size_t)Bsz * 3 * 64 * 64;            // 98304
    float* tmp_re  = alloc(IMG);
    float* tmp_im  = alloc(IMG);
    float* amp_raw = alloc(IMG);
    float* pha_raw = alloc(IMG);
    float* mid     = alloc((size_t)Bsz * 8 * 64 * 64);       // 262144
    float* ampc    = alloc(IMG);
    float* phac    = alloc(IMG);
    float* xz_a    = alloc((size_t)Bsz * Lseq * 12);
    float* xz_p    = alloc((size_t)Bsz * Lseq * 12);
    float* xc_a    = alloc((size_t)Bsz * Lseq * DIm);
    float* xc_p    = alloc((size_t)Bsz * Lseq * DIm);
    float* xdbl_a  = alloc((size_t)Bsz * Lseq * 33);
    float* xdbl_p  = alloc((size_t)Bsz * Lseq * 33);
    float* dt_a    = alloc((size_t)Bsz * Lseq * DIm);
    float* dt_p    = alloc((size_t)Bsz * Lseq * DIm);
    float* ys_a    = alloc((size_t)Bsz * Lseq * DIm);
    float* ys_p    = alloc((size_t)Bsz * Lseq * DIm);
    float* mo_a    = alloc(IMG);
    float* mo_p    = alloc(IMG);
    float* sm_a    = alloc(IMG);
    float* sm_p    = alloc(IMG);
    float* Y_a     = alloc((size_t)Lseq * NPAD);
    float* Y_p     = alloc((size_t)Lseq * NPAD);
    float* Y_s     = alloc((size_t)Lseq * NPAD);
    float* cross   = alloc(IMG);
    float* aout    = alloc(IMG);
    float* pout    = alloc(IMG);
    bf16_t* Xp_a = (bf16_t*)alloc((size_t)NPAD * KTOT / 2);
    bf16_t* Xp_p = (bf16_t*)alloc((size_t)NPAD * KTOT / 2);
    bf16_t* Xp_s = (bf16_t*)alloc((size_t)NPAD * KTOT / 2);
    (void)ws_size; (void)n_in; (void)in_sizes; (void)out_size;

    const int BL = Bsz * Lseq;
    const int LINES = Bsz * 3 * 64;                          // 1536

    // forward fft2
    k_dft64<<<LINES, 64, 0, stream>>>(x, nullptr, tmp_re, tmp_im, 0, -1.f, 1.f, 0, 1);
    k_dft64<<<LINES, 64, 0, stream>>>(tmp_re, tmp_im, amp_raw, pha_raw, 1, -1.f, 1.f, 1, 1);

    // convb + residual, both branches
    k_convb1<<<cdiv((int)(Bsz * 8 * 4096), 256), 256, 0, stream>>>(amp_raw, cb1_w, cb1_b, mid, Bsz * 8 * 4096);
    k_convb2<<<cdiv((int)IMG, 256), 256, 0, stream>>>(mid, cb2_w, cb2_b, amp_raw, ampc, (int)IMG);
    k_convb1<<<cdiv((int)(Bsz * 8 * 4096), 256), 256, 0, stream>>>(pha_raw, cb1_w, cb1_b, mid, Bsz * 8 * 4096);
    k_convb2<<<cdiv((int)IMG, 256), 256, 0, stream>>>(mid, cb2_w, cb2_b, pha_raw, phac, (int)IMG);

    // layernorm (in place)
    k_ln<<<cdiv(BL, 256), 256, 0, stream>>>(ampc, ln_g, ln_b, BL);
    k_ln<<<cdiv(BL, 256), 256, 0, stream>>>(phac, ln_g, ln_b, BL);

    // mamba branches
    run_mamba(ampc, m1p[0], m1p[1], m1p[2], m1p[3], m1p[4], m1p[5], m1p[6], m1p[7], m1p[8],
              xz_a, xc_a, xdbl_a, dt_a, ys_a, mo_a, stream);
    run_mamba(phac, m2p[0], m2p[1], m2p[2], m2p[3], m2p[4], m2p[5], m2p[6], m2p[7], m2p[8],
              xz_p, xc_p, xdbl_p, dt_p, ys_p, mo_p, stream);

    // batch softmax
    k_softmax_b<<<cdiv(12288, 256), 256, 0, stream>>>(mo_a, sm_a, 12288);
    k_softmax_b<<<cdiv(12288, 256), 256, 0, stream>>>(mo_p, sm_p, 12288);

    // pack activations to bf16
    k_pack<<<cdiv(NPAD * KTOT, 256), 256, 0, stream>>>(mo_a, Xp_a, NPAD * KTOT);
    k_pack<<<cdiv(NPAD * KTOT, 256), 256, 0, stream>>>(mo_p, Xp_p, NPAD * KTOT);
    k_pack<<<cdiv(NPAD * KTOT, 256), 256, 0, stream>>>(sm_a, Xp_s, NPAD * KTOT);

    // three weight-bound WMMA GEMMs
    k_gemm_wmma<<<Lseq / 64, 128, 0, stream>>>(c11_w, Xp_a, Y_a);
    k_gemm_wmma<<<Lseq / 64, 128, 0, stream>>>(c12_w, Xp_p, Y_p);
    k_gemm_wmma<<<Lseq / 64, 128, 0, stream>>>(cr1_w, Xp_s, Y_s);

    // cross term (amp3*amp4, shared by both outputs) and final compose
    k_cross<<<cdiv(12288, 256), 256, 0, stream>>>(Y_s, cr1_b, cross, 12288);
    k_compose<<<cdiv((int)(Bsz * 12288), 256), 256, 0, stream>>>(
        Y_a, c11_b, sm_a, Y_p, c12_b, sm_p, cross, amp_raw, pha_raw, aout, pout, Bsz * 12288);

    // inverse fft2, real part to d_out
    k_dft64<<<LINES, 64, 0, stream>>>(aout, pout, tmp_re, tmp_im, 0, 1.f, 1.f / 64.f, 1, 1);
    k_dft64<<<LINES, 64, 0, stream>>>(tmp_re, tmp_im, (float*)d_out, nullptr, 1, 1.f, 1.f / 64.f, 1, 0);
}
